// SelfAttention_35424890258012
// MI455X (gfx1250) — matrix operand order repaired and checked by reading, not compiled, not run on hardware
//
#include <hip/hip_runtime.h>

typedef __attribute__((ext_vector_type(16))) _Float16 v16h;
typedef __attribute__((ext_vector_type(8)))  _Float16 v8h;
typedef __attribute__((ext_vector_type(8)))  float    v8f;

#define B_  2
#define T_  2048
#define C_  1024
#define H_  16
#define D_  64
#define BT_ (B_ * T_)

union AFrag { v16h v; v8h h[2]; };

__device__ __forceinline__ v8f wmma_f16(v16h a, v16h b, v8f c) {
  // D = A(16x32 f16) * B(32x16 f16) + C(16x16 f32)
  return __builtin_amdgcn_wmma_f32_16x16x32_f16(
      /*neg_a=*/false, a, /*neg_b=*/false, b,
      /*c_mod=*/(short)0, c, /*reuse_a=*/false, /*reuse_b=*/false);
}

// ---------------- fp32 -> f16 convert (x) ----------------
__global__ void cvt_x_kernel(const float* __restrict__ x,
                             _Float16* __restrict__ xh, int n) {
  int i = blockIdx.x * blockDim.x + threadIdx.x;
  if (i < n) xh[i] = (_Float16)x[i];
}

// ---- pack W (fp32 [N_out=1024][K_in=1024]) into B-fragment-native f16 tiles:
// dst[((n>>4)*(C/16) + (k>>4))*256 + (n&15)*16 + (k&15)]
// so a WMMA B fragment (16 cols x 32 k) is a contiguous 32B load per lane.
__global__ void pack_w_kernel(const float* __restrict__ w,
                              _Float16* __restrict__ wpk) {
  int i = blockIdx.x * blockDim.x + threadIdx.x;
  if (i >= C_ * C_) return;
  int nrow = i >> 10;      // output feature (row of W)
  int k    = i & (C_ - 1); // input feature
  int dst = (((nrow >> 4) * (C_ / 16) + (k >> 4)) << 8) + ((nrow & 15) << 4) + (k & 15);
  wpk[dst] = (_Float16)w[i];
}

// ---------------- generic 16x64-per-wave GEMM: Y = X @ W^T + b ----------------
// mode 0: store f16 Q*(1/sqrt(D)) as [BH][T][D]
// mode 1: store f16 K            as [BH][T][D]
// mode 2: store f16 V transposed as [BH][D][T]
// mode 3: store f32 Y            as [BT][C] (final output)
__global__ void __launch_bounds__(32)
gemm16_kernel(const _Float16* __restrict__ X, const _Float16* __restrict__ Wpk,
              const float* __restrict__ bias, _Float16* __restrict__ out16,
              float* __restrict__ out32, int mode) {
  const int lane = threadIdx.x & 31;
  const int hs = lane >> 4;   // lane-half select
  const int lm = lane & 15;
  const int mbase = blockIdx.y * 16;
  const int nbase = blockIdx.x * 64;

  v8f acc[4] = {};
  const _Float16* xrow = X + (size_t)(mbase + lm) * C_;

  for (int k0 = 0; k0 < C_; k0 += 32) {
    AFrag a;                                    // A: lane=M, K contiguous 16 per half
    a.h[0] = *(const v8h*)(xrow + k0 + 16 * hs);
    a.h[1] = *(const v8h*)(xrow + k0 + 16 * hs + 8);
#pragma unroll
    for (int nt = 0; nt < 4; ++nt) {
      const _Float16* bp = Wpk +
          ((((size_t)((nbase >> 4) + nt)) * (C_ / 16) + (k0 >> 4) + hs) << 8) +
          (lm << 4);
      v16h b = *(const v16h*)bp;                // contiguous 32B per lane
      acc[nt] = wmma_f16(a.v, b, acc[nt]);
    }
  }

#pragma unroll
  for (int nt = 0; nt < 4; ++nt) {
    int col = nbase + nt * 16 + lm;
    float bv = bias[col];
#pragma unroll
    for (int r = 0; r < 8; ++r) {
      int row = mbase + r + 8 * hs;             // C-layout: M = r + 8*half
      float val = acc[nt][r] + bv;
      if (mode <= 1) {
        if (mode == 0) val *= 0.125f;           // 1/sqrt(64) folded into Q
        int b = row >> 11, t = row & (T_ - 1);
        int h = col >> 6,  d = col & 63;
        out16[((((size_t)b * H_ + h) * T_ + t) << 6) + d] = (_Float16)val;
      } else if (mode == 2) {                   // V stored transposed [BH][D][T]
        int b = row >> 11, t = row & (T_ - 1);
        int h = col >> 6,  d = col & 63;
        out16[((((size_t)b * H_ + h) * D_ + d) << 11) + t] = (_Float16)val;
      } else {
        out32[(size_t)row * C_ + col] = val;
      }
    }
  }
}

// ---------------- flash attention: 1 wave per 32-query tile ----------------
// Two 16-row M-tiles per wave share every K/V fragment load:
// per 32-key block: 8 S-WMMAs + 8 PV-WMMAs against 8 K/V loads.
__global__ void __launch_bounds__(32)
attn_kernel(const _Float16* __restrict__ Q, const _Float16* __restrict__ K,
            const _Float16* __restrict__ Vt, const int* __restrict__ mask,
            _Float16* __restrict__ att) {
  __shared__ _Float16 pS[32 * 32];              // P relayout buffer (2 KB)

  const int lane = threadIdx.x & 31;
  const int hs = lane >> 4;
  const int lm = lane & 15;
  const int bh = blockIdx.y;                    // b*H + h
  const int qbase = blockIdx.x * 32;
  const int b = bh >> 4, h = bh & 15;

  const _Float16* Qh = Q  + (size_t)bh * T_ * D_;
  const _Float16* Kh = K  + (size_t)bh * T_ * D_;
  const _Float16* Vh = Vt + (size_t)bh * D_ * T_;
  const int* mk = mask + (size_t)b * T_;

  // Q A-fragments for both M-tiles (reused all loop): K-dim = d
  AFrag qa[2][2];
#pragma unroll
  for (int t = 0; t < 2; ++t) {
    const _Float16* qrow = Qh + (size_t)(qbase + t * 16 + lm) * D_;
    qa[t][0].h[0] = *(const v8h*)(qrow + 16 * hs);
    qa[t][0].h[1] = *(const v8h*)(qrow + 16 * hs + 8);
    qa[t][1].h[0] = *(const v8h*)(qrow + 32 + 16 * hs);
    qa[t][1].h[1] = *(const v8h*)(qrow + 32 + 16 * hs + 8);
  }

  v8f accO[2][4] = {};
  float mrow[2][8], lrow[2][8];
#pragma unroll
  for (int t = 0; t < 2; ++t)
#pragma unroll
    for (int r = 0; r < 8; ++r) { mrow[t][r] = -__builtin_inff(); lrow[t][r] = 0.f; }

  for (int key0 = 0; key0 < T_; key0 += 32) {
    // prefetch next key block's K rows and V columns into caches
    if (key0 + 32 < T_) {
      __builtin_prefetch(Kh + (size_t)(key0 + 32 + lane) * D_, 0, 1);
      __builtin_prefetch(Vh + (size_t)lane * T_ + key0 + 32, 0, 1);
      __builtin_prefetch(Vh + (size_t)(32 + lane) * T_ + key0 + 32, 0, 1);
    }

    // ---- K^T B-fragments (shared by both M-tiles)
    v16h b00 = *(const v16h*)(Kh + (size_t)(key0 + lm) * D_ + 16 * hs);
    v16h b01 = *(const v16h*)(Kh + (size_t)(key0 + lm) * D_ + 32 + 16 * hs);
    v16h b10 = *(const v16h*)(Kh + (size_t)(key0 + 16 + lm) * D_ + 16 * hs);
    v16h b11 = *(const v16h*)(Kh + (size_t)(key0 + 16 + lm) * D_ + 32 + 16 * hs);

    v8f s[2][2] = {};
#pragma unroll
    for (int t = 0; t < 2; ++t) {
      s[t][0] = wmma_f16(qa[t][0].v, b00, s[t][0]);
      s[t][0] = wmma_f16(qa[t][1].v, b01, s[t][0]);
      s[t][1] = wmma_f16(qa[t][0].v, b10, s[t][1]);
      s[t][1] = wmma_f16(qa[t][1].v, b11, s[t][1]);
    }

    const int m0 = mk[key0 + lm];
    const int m1 = mk[key0 + 16 + lm];

    float corr[2][8];
#pragma unroll
    for (int t = 0; t < 2; ++t) {
      float sv0[8], sv1[8], bmax[8];
#pragma unroll
      for (int r = 0; r < 8; ++r) {
        float e0 = m0 ? (float)s[t][0][r] : -1e30f;   // masked_softmax fill
        float e1 = m1 ? (float)s[t][1][r] : -1e30f;
        sv0[r] = e0; sv1[r] = e1;
        bmax[r] = fmaxf(e0, e1);
      }
      // row max: each row's 16 columns live in one 16-lane half -> xor reduce
#pragma unroll
      for (int r = 0; r < 8; ++r) {
#pragma unroll
        for (int m = 1; m <= 8; m <<= 1)
          bmax[r] = fmaxf(bmax[r], __shfl_xor(bmax[r], m, 32));
      }

      float psum[8];
#pragma unroll
      for (int r = 0; r < 8; ++r) {
        float mnew = fmaxf(mrow[t][r], bmax[r]);
        corr[t][r] = __expf(mrow[t][r] - mnew);
        mrow[t][r] = mnew;
        float p0 = __expf(sv0[r] - mnew);
        float p1 = __expf(sv1[r] - mnew);
        psum[r] = p0 + p1;
        // stash P row-major f16 (32 rows x 32 keys) for C->A relayout
        pS[(t * 16 + r + 8 * hs) * 32 + lm]      = (_Float16)p0;
        pS[(t * 16 + r + 8 * hs) * 32 + 16 + lm] = (_Float16)p1;
      }
#pragma unroll
      for (int r = 0; r < 8; ++r) {
#pragma unroll
        for (int m = 1; m <= 8; m <<= 1)
          psum[r] += __shfl_xor(psum[r], m, 32);
        lrow[t][r] = lrow[t][r] * corr[t][r] + psum[r];
      }
#pragma unroll
      for (int nt = 0; nt < 4; ++nt)
#pragma unroll
        for (int r = 0; r < 8; ++r) accO[t][nt][r] *= corr[t][r];
    }

    __syncthreads();
    // P as A fragments: lane=query row, halves K(key) contiguous 16 per half
    AFrag pa[2];
#pragma unroll
    for (int t = 0; t < 2; ++t) {
      pa[t].h[0] = *(const v8h*)(&pS[(t * 16 + lm) * 32 + 16 * hs]);
      pa[t].h[1] = *(const v8h*)(&pS[(t * 16 + lm) * 32 + 16 * hs + 8]);
    }

    // O += P (16x32) @ V (32x64): each V fragment feeds both M-tiles
#pragma unroll
    for (int nt = 0; nt < 4; ++nt) {
      v16h bv = *(const v16h*)(Vh + (size_t)(nt * 16 + lm) * T_ + key0 + 16 * hs);
#pragma unroll
      for (int t = 0; t < 2; ++t)
        accO[t][nt] = wmma_f16(pa[t].v, bv, accO[t][nt]);
    }
    __syncthreads();
  }

#pragma unroll
  for (int t = 0; t < 2; ++t) {
#pragma unroll
    for (int r = 0; r < 8; ++r) {
      float inv = 1.f / lrow[t][r];
#pragma unroll
      for (int nt = 0; nt < 4; ++nt) {
        int row = qbase + t * 16 + r + 8 * hs;
        int col = h * 64 + nt * 16 + lm;
        att[((size_t)b * T_ + row) * C_ + col] = (_Float16)(accO[t][nt][r] * inv);
      }
    }
  }
}

// ---------------- host side ----------------
extern "C" void kernel_launch(void* const* d_in, const int* in_sizes, int n_in,
                              void* d_out, int out_size, void* d_ws, size_t ws_size,
                              hipStream_t stream) {
  const float* x  = (const float*)d_in[0];
  const int*   mk = (const int*)  d_in[1];
  const float* Wq = (const float*)d_in[2];
  const float* bq = (const float*)d_in[3];
  const float* Wk = (const float*)d_in[4];
  const float* bk = (const float*)d_in[5];
  const float* Wv = (const float*)d_in[6];
  const float* bv = (const float*)d_in[7];
  const float* Wp = (const float*)d_in[8];
  const float* bp = (const float*)d_in[9];
  float* out = (float*)d_out;

  char* ws = (char*)d_ws;
  size_t off = 0;
  auto carve = [&](size_t bytes) {
    char* p = ws + off;
    off = (off + bytes + 255) & ~(size_t)255;
    return p;
  };
  _Float16* xh  = (_Float16*)carve((size_t)BT_ * C_ * 2);   // x in f16
  _Float16* wqp = (_Float16*)carve((size_t)C_ * C_ * 2);    // packed weights
  _Float16* wkp = (_Float16*)carve((size_t)C_ * C_ * 2);
  _Float16* wvp = (_Float16*)carve((size_t)C_ * C_ * 2);
  _Float16* wpp = (_Float16*)carve((size_t)C_ * C_ * 2);
  _Float16* qh  = (_Float16*)carve((size_t)BT_ * C_ * 2);   // [BH][T][D]
  _Float16* kh  = (_Float16*)carve((size_t)BT_ * C_ * 2);   // [BH][T][D]
  _Float16* vt  = (_Float16*)carve((size_t)BT_ * C_ * 2);   // [BH][D][T]
  _Float16* att = (_Float16*)carve((size_t)BT_ * C_ * 2);   // [BT][C]

  const int nX = BT_ * C_;
  cvt_x_kernel<<<(nX + 255) / 256, 256, 0, stream>>>(x, xh, nX);
  const int nW = C_ * C_;
  pack_w_kernel<<<(nW + 255) / 256, 256, 0, stream>>>(Wq, wqp);
  pack_w_kernel<<<(nW + 255) / 256, 256, 0, stream>>>(Wk, wkp);
  pack_w_kernel<<<(nW + 255) / 256, 256, 0, stream>>>(Wv, wvp);
  pack_w_kernel<<<(nW + 255) / 256, 256, 0, stream>>>(Wp, wpp);

  dim3 g(C_ / 64, BT_ / 16);
  gemm16_kernel<<<g, 32, 0, stream>>>(xh, wqp, bq, qh, nullptr, 0);
  gemm16_kernel<<<g, 32, 0, stream>>>(xh, wkp, bk, kh, nullptr, 1);
  gemm16_kernel<<<g, 32, 0, stream>>>(xh, wvp, bv, vt, nullptr, 2);

  attn_kernel<<<dim3(T_ / 32, B_ * H_), 32, 0, stream>>>(qh, kh, vt, mk, att);

  gemm16_kernel<<<g, 32, 0, stream>>>(att, wpp, bp, nullptr, out, 3);
}
